// BigModel_3539053052154
// MI455X (gfx1250) — compile-verified, matrix operand order checked by reading, not run on hardware
//
#include <hip/hip_runtime.h>
#include <hip/hip_bf16.h>

typedef __attribute__((ext_vector_type(16))) _Float16 v16h;
typedef __attribute__((ext_vector_type(8)))  float    v8f;
typedef __attribute__((ext_vector_type(4)))  unsigned int v4u;
typedef int async_i4 __attribute__((__vector_size__(16)));

#define NMEL 80
#define TLEN 800
#define BSZ  16
#define LSEQ 256
#define DMOD 256
#define HIDD 512
#define SDM  384

// ---- async global->LDS copy (16 bytes), with portable fallback ----
__device__ __forceinline__ void async_cp16(const _Float16* g, _Float16* l) {
#if __has_builtin(__builtin_amdgcn_global_load_async_to_lds_b128)
  __builtin_amdgcn_global_load_async_to_lds_b128(
      (__attribute__((address_space(1))) async_i4*)g,
      (__attribute__((address_space(3))) async_i4*)l, 0, 0);
#else
  *(v4u*)l = *(const v4u*)g;
#endif
}

__device__ __forceinline__ void wait_async() {
#if __has_builtin(__builtin_amdgcn_s_wait_asynccnt)
  __builtin_amdgcn_s_wait_asynccnt(0);
#elif __has_builtin(__builtin_amdgcn_global_load_async_to_lds_b128)
  asm volatile("s_wait_asynccnt 0x0" ::: "memory");
#endif
}

// ---- WMMA fragment helper (16x16x32 f16, wave32) ----
// lane holds row (lane&15); element i covers K = i + 8*(lane>=16) + 8*(i>=8)
// => two contiguous 8-half (16B) runs per lane: vectorizes to b128 loads.
__device__ __forceinline__ int kofs(int i, int hi) {
  return i + (hi << 3) + ((i >= 8) << 3);
}

__device__ __forceinline__ v16h frag_f16_rowK(const _Float16* base, int ld,
                                              int r0, int k0, int lane) {
  int r = r0 + (lane & 15);
  int hi = (lane >> 4) & 1;
  const _Float16* rp = base + (long long)r * ld;
  v16h f;
#pragma unroll
  for (int i = 0; i < 16; ++i)
    f[i] = rp[k0 + kofs(i, hi)];
  return f;
}

// C/D tile store: lane col = n0+(lane&15); rows m0 + j + 8*(lane>=16).
// EPI: 0 plain(+bias), 1 relu, 2 mel-layout ([B,NM,T], row = t*16+b).
template<int EPI>
__device__ __forceinline__ void store_tile(v8f acc, float* Cb, _Float16* Chb,
                                           const float* bias, int m0, int n0,
                                           int M, int N, int ldc, int lane) {
  int col = n0 + (lane & 15);
  int hi = (lane >> 4) & 1;
  float bv = (bias != nullptr && col < N) ? bias[col] : 0.0f;
#pragma unroll
  for (int j = 0; j < 8; ++j) {
    int row = m0 + j + (hi << 3);
    if (row < M && col < N) {
      float v = acc[j] + bv;
      if (EPI == 1) v = fmaxf(v, 0.0f);
      if (EPI == 2) {
        Cb[((long long)(row & 15) * NMEL + col) * TLEN + (row >> 4)] = v;
      } else {
        if (Cb)  Cb[(long long)row * ldc + col] = v;
        if (Chb) Chb[(long long)row * ldc + col] = (_Float16)v;
      }
    }
  }
}

// ---- Block-cooperative LDS-staged f16 WMMA GEMM ----
// Macro-tile 128x64 per block (8 waves as 4x2, each wave 32x32 = 4 WMMA tiles).
// A [M,K] f16 row-major, B [N,K] f16 row-major (x @ w.T form).
// Requires M%128==0, K%32==0, N%16==0 (N<64-multiples handled by clamp+guards).
template<int EPI>
__global__ void __launch_bounds__(256)
gemm16(const _Float16* __restrict__ A, const _Float16* __restrict__ B,
       const float* __restrict__ bias, float* __restrict__ C,
       _Float16* __restrict__ Ch, int M, int N, int K,
       int lda, int ldb, int ldc, int nInner,
       long long sAo, long long sAi, long long sBo, long long sBi,
       long long sCo, long long sCi) {
  __shared__ __align__(16) _Float16 As[128 * 32];  // 8 KB
  __shared__ __align__(16) _Float16 Bs[64 * 32];   // 4 KB
  int bo = blockIdx.y / nInner, bi = blockIdx.y % nInner;
  const _Float16* Ab = A + bo * sAo + bi * sAi;
  const _Float16* Bb = B + bo * sBo + bi * sBi;
  float* Cb = C ? (C + bo * sCo + bi * sCi) : nullptr;
  _Float16* Chb = Ch ? (Ch + bo * sCo + bi * sCi) : nullptr;
  int tid = threadIdx.x;
  int lane = tid & 31;
  int wid = tid >> 5;               // 0..7
  int mw = (wid >> 1) << 5;         // wave M offset in macro-tile: 0,32,64,96
  int nw2 = (wid & 1) << 5;         // wave N offset: 0,32
  int arow = tid >> 2;              // 0..63
  int aq = (tid & 3) << 3;          // 0,8,16,24 (halfs)
  int tMB = M >> 7;
  int tNB = (N + 63) >> 6;
  for (int s = blockIdx.x; s < tMB * tNB; s += gridDim.x) {
    int sm = s / tNB, sn = s % tNB;
    int m0 = sm << 7, n0 = sn << 6;
    v8f a00 = {}, a01 = {}, a10 = {}, a11 = {};
    for (int k0 = 0; k0 < K; k0 += 32) {
      // cooperative stage: A panel 128x32 (2 rows/thread), B panel 64x32 (1 row)
      int rb = n0 + arow;
      if (rb > N - 1) rb = N - 1;  // clamp (dup rows, stores guarded)
      async_cp16(Ab + (long long)(m0 + arow) * lda + k0 + aq, &As[arow * 32 + aq]);
      async_cp16(Ab + (long long)(m0 + arow + 64) * lda + k0 + aq, &As[(arow + 64) * 32 + aq]);
      async_cp16(Bb + (long long)rb * ldb + k0 + aq, &Bs[arow * 32 + aq]);
      wait_async();
      __syncthreads();
      v16h af0 = frag_f16_rowK(As, 32, mw, 0, lane);
      v16h af1 = frag_f16_rowK(As, 32, mw + 16, 0, lane);
      v16h bf0 = frag_f16_rowK(Bs, 32, nw2, 0, lane);
      v16h bf1 = frag_f16_rowK(Bs, 32, nw2 + 16, 0, lane);
      a00 = __builtin_amdgcn_wmma_f32_16x16x32_f16(false, af0, false, bf0, (short)0, a00, false, false);
      a01 = __builtin_amdgcn_wmma_f32_16x16x32_f16(false, af0, false, bf1, (short)0, a01, false, false);
      a10 = __builtin_amdgcn_wmma_f32_16x16x32_f16(false, af1, false, bf0, (short)0, a10, false, false);
      a11 = __builtin_amdgcn_wmma_f32_16x16x32_f16(false, af1, false, bf1, (short)0, a11, false, false);
      __syncthreads();
    }
    store_tile<EPI>(a00, Cb, Chb, bias, m0 + mw, n0 + nw2, M, N, ldc, lane);
    store_tile<EPI>(a01, Cb, Chb, bias, m0 + mw, n0 + nw2 + 16, M, N, ldc, lane);
    store_tile<EPI>(a10, Cb, Chb, bias, m0 + mw + 16, n0 + nw2, M, N, ldc, lane);
    store_tile<EPI>(a11, Cb, Chb, bias, m0 + mw + 16, n0 + nw2 + 16, M, N, ldc, lane);
  }
}

// ---- f32 -> f16 conversions ----
__global__ void k_f2h(const float* __restrict__ s, _Float16* __restrict__ d, long long n) {
  long long i = (long long)blockIdx.x * blockDim.x + threadIdx.x;
  if (i < n) d[i] = (_Float16)s[i];
}

__global__ void k_f2h_pad(const float* __restrict__ s, _Float16* __restrict__ d,
                          int rows, int Ks, int Kd) {
  long long i = (long long)blockIdx.x * blockDim.x + threadIdx.x;
  if (i >= (long long)rows * Kd) return;
  int k = (int)(i % Kd);
  long long r = i / Kd;
  d[i] = (_Float16)((k < Ks) ? s[r * Ks + k] : 0.0f);
}

// ---- Reference encoder convs ----
__global__ void k_conv1(const float* __restrict__ mel, const float* __restrict__ w,
                        const float* __restrict__ bias, float* __restrict__ out, int kw) {
  long long idx = (long long)blockIdx.x * blockDim.x + threadIdx.x;
  if (idx >= (long long)BSZ * 128 * TLEN) return;
  int t = (int)(idx % TLEN);
  int o = (int)((idx / TLEN) % 128);
  int b = (int)(idx / ((long long)TLEN * 128));
  int p = kw >> 1;
  float s = bias[o];
  for (int m = 0; m < NMEL; ++m) {
    const float* mp = mel + ((long long)b * NMEL + m) * TLEN;
    const float* wp = w + ((long long)o * NMEL + m) * kw;
    for (int d = 0; d < kw; ++d) {
      int tt = t + d - p;
      if ((unsigned)tt < (unsigned)TLEN) s += mp[tt] * wp[d];
    }
  }
  out[idx] = s;  // no relu after conv1 (matches reference)
}

__global__ void k_conv(const float* __restrict__ x, const float* __restrict__ w,
                       const float* __restrict__ bias, float* __restrict__ out,
                       int Cin, int Cout, int kw) {
  long long idx = (long long)blockIdx.x * blockDim.x + threadIdx.x;
  if (idx >= (long long)BSZ * Cout * TLEN) return;
  int t = (int)(idx % TLEN);
  int o = (int)((idx / TLEN) % Cout);
  int b = (int)(idx / ((long long)TLEN * Cout));
  int p = kw >> 1;
  float s = bias[o];
  for (int i = 0; i < Cin; ++i) {
    const float* xp = x + ((long long)b * Cin + i) * TLEN;
    const float* wp = w + ((long long)o * Cin + i) * kw;
    for (int d = 0; d < kw; ++d) {
      int tt = t + d - p;
      if ((unsigned)tt < (unsigned)TLEN) s += xp[tt] * wp[d];
    }
  }
  out[idx] = fmaxf(s, 0.0f);
}

__global__ void k_pool_fc(const float* __restrict__ x3, const float* __restrict__ fcw,
                          const float* __restrict__ fcb, const float* __restrict__ clsw,
                          const float* __restrict__ clsb, float* __restrict__ style,
                          float* __restrict__ logits, int outDim, int clsDim,
                          int styleOff, int domax) {
  __shared__ float pooled[256];
  __shared__ float feat[256];
  int b = blockIdx.x, c = threadIdx.x;
  const float* xp = x3 + ((long long)b * 256 + c) * TLEN;
  float s = domax ? -1e30f : 0.0f;
  for (int t = 0; t < TLEN; ++t) {
    float v = xp[t];
    s = domax ? fmaxf(s, v) : (s + v);
  }
  pooled[c] = domax ? s : (s * (1.0f / TLEN));
  __syncthreads();
  if (c < outDim) {
    float e = fcb[c];
    for (int i = 0; i < 256; ++i) e += pooled[i] * fcw[c * 256 + i];
    feat[c] = e;
    style[b * SDM + styleOff + c] = e;
  }
  __syncthreads();
  if (c < clsDim) {
    float lg = clsb[c];
    for (int i = 0; i < outDim; ++i) lg += feat[i] * clsw[c * outDim + i];
    logits[b * clsDim + c] = lg;
  }
}

// ---- Transformer pieces ----
__global__ void k_embed_pe(const int* __restrict__ tok, const float* __restrict__ emb,
                           float* __restrict__ x, _Float16* __restrict__ xh,
                           int* __restrict__ padm) {
  int idx = blockIdx.x * blockDim.x + threadIdx.x;
  if (idx >= BSZ * LSEQ * DMOD) return;
  int d = idx & (DMOD - 1);
  int bl = idx >> 8;
  int l = bl & (LSEQ - 1);
  int tkn = tok[bl];
  int i2 = d >> 1;
  float div = __expf((float)i2 * (-2.0f * 9.210340371976184f / (float)DMOD));
  float ang = (float)l * div;
  float pe = (d & 1) ? __cosf(ang) : __sinf(ang);
  float v = emb[tkn * DMOD + d] + pe;
  x[idx] = v;
  xh[idx] = (_Float16)v;
  if (d == 0) padm[bl] = (tkn == 0) ? 1 : 0;
}

__global__ void k_softmax_mask(_Float16* __restrict__ s, const int* __restrict__ padm) {
  __shared__ float red[256];
  int row = blockIdx.x;          // (b*4+h)*256 + i
  int b = row >> 10;
  int j = threadIdx.x;
  _Float16* rp = s + (long long)row * 256;
  float v = (float)rp[j] * 0.125f;  // 1/sqrt(64)
  if (padm[b * LSEQ + j]) v = -1e9f;
  red[j] = v;
  __syncthreads();
  for (int o = 128; o > 0; o >>= 1) {
    if (j < o) red[j] = fmaxf(red[j], red[j + o]);
    __syncthreads();
  }
  float mx = red[0];
  __syncthreads();
  float e = __expf(v - mx);
  red[j] = e;
  __syncthreads();
  for (int o = 128; o > 0; o >>= 1) {
    if (j < o) red[j] += red[j + o];
    __syncthreads();
  }
  rp[j] = (_Float16)(e / red[0]);
}

// vT[(b*4+h)*64 + d][j] = v[b, j, h*64+d]  (from qkv f16, v slice)
__global__ void k_tr_v(const _Float16* __restrict__ qkvh, _Float16* __restrict__ vT) {
  int idx = blockIdx.x * blockDim.x + threadIdx.x;
  if (idx >= 64 * 64 * 256) return;
  int j = idx & 255;
  int d = (idx >> 8) & 63;
  int bh = idx >> 14;
  int b = bh >> 2, h = bh & 3;
  vT[idx] = qkvh[((long long)(b * 256 + j)) * 768 + 512 + h * 64 + d];
}

__global__ void k_add_ln(float* __restrict__ x, const float* __restrict__ r,
                         const float* __restrict__ g, const float* __restrict__ bta,
                         _Float16* __restrict__ xh) {
  __shared__ float red[256];
  int row = blockIdx.x, d = threadIdx.x;
  float v = x[row * 256 + d] + r[row * 256 + d];
  red[d] = v;
  __syncthreads();
  for (int o = 128; o > 0; o >>= 1) {
    if (d < o) red[d] += red[d + o];
    __syncthreads();
  }
  float mean = red[0] * (1.0f / 256.0f);
  __syncthreads();
  float c = v - mean;
  red[d] = c * c;
  __syncthreads();
  for (int o = 128; o > 0; o >>= 1) {
    if (d < o) red[d] += red[d + o];
    __syncthreads();
  }
  float var = red[0] * (1.0f / 256.0f);
  float y = c * rsqrtf(var + 1e-5f) * g[d] + bta[d];
  x[row * 256 + d] = y;
  xh[row * 256 + d] = (_Float16)y;
}

// ---- Decoder ----
// xdec f16, K padded 976 -> 1024 with zeros
__global__ void k_build_xdec(const float* __restrict__ enc, const float* __restrict__ style,
                             const float* __restrict__ mel, _Float16* __restrict__ xd) {
  long long idx = (long long)blockIdx.x * blockDim.x + threadIdx.x;
  if (idx >= (long long)TLEN * BSZ * 1024) return;
  int j = (int)(idx & 1023);
  int row = (int)(idx >> 10);
  int b = row & 15;
  int t = row >> 4;
  float v;
  if (j < HIDD) {
    int l = (t < LSEQ) ? t : (LSEQ - 1);
    v = enc[((long long)b * LSEQ + l) * HIDD + j];
  } else if (j < HIDD + SDM) {
    v = style[b * SDM + (j - HIDD)];
  } else if (j < HIDD + SDM + NMEL) {
    int c = j - HIDD - SDM;
    v = (t == 0) ? 0.0f : mel[((long long)b * NMEL + c) * TLEN + (t - 1)];
  } else {
    v = 0.0f;
  }
  xd[idx] = (_Float16)v;
}

__device__ __forceinline__ float sigm(float x) { return 1.0f / (1.0f + __expf(-x)); }

// Persistent 2-layer LSTM scan: 1 workgroup, 32 waves, 800 steps.
__global__ void __launch_bounds__(1024)
k_scan(const float* __restrict__ gx1, const _Float16* __restrict__ whh1h,
       const _Float16* __restrict__ wih2h, const _Float16* __restrict__ whh2h,
       const float* __restrict__ bhh1, const float* __restrict__ bih2,
       const float* __restrict__ bhh2, _Float16* __restrict__ H2) {
  __shared__ _Float16 Gs[16 * 2048];
  __shared__ _Float16 h1h[16 * HIDD];
  __shared__ _Float16 h2h[16 * HIDD];
  int tid = threadIdx.x;
  int lane = tid & 31;
  int wid = tid >> 5;
  float c1r[8], c2r[8];
#pragma unroll
  for (int q = 0; q < 8; ++q) { c1r[q] = 0.0f; c2r[q] = 0.0f; }
  for (int e = tid; e < 16 * HIDD; e += 1024) {
    h1h[e] = (_Float16)0.0f;
    h2h[e] = (_Float16)0.0f;
  }
  __syncthreads();

  for (int t = 0; t < TLEN; ++t) {
    const float* g1p = gx1 + (long long)t * 16 * 2048;
    // phase 1: G = gx1[t] + h1 @ whh1.T + bhh1
    for (int nt = wid; nt < 128; nt += 32) {
      v8f acc = {};
      for (int k0 = 0; k0 < HIDD; k0 += 32) {
        v16h a = frag_f16_rowK(h1h, HIDD, 0, k0, lane);
        v16h b = frag_f16_rowK(whh1h, HIDD, nt * 16, k0, lane);
        acc = __builtin_amdgcn_wmma_f32_16x16x32_f16(false, a, false, b,
                                                     (short)0, acc, false, false);
      }
      int col = nt * 16 + (lane & 15);
      int hi = (lane >> 4) & 1;
      float bv = bhh1[col];
#pragma unroll
      for (int j = 0; j < 8; ++j) {
        int bb = j + (hi << 3);
        Gs[bb * 2048 + col] = (_Float16)(acc[j] + g1p[bb * 2048 + col] + bv);
      }
    }
    __syncthreads();
    // phase 2: LSTM cell 1 (gate order i,f,g,o)
#pragma unroll
    for (int q = 0; q < 8; ++q) {
      int e = (tid << 3) + q;
      int bb = e >> 9, u = e & 511;
      const int gb = bb * 2048;
      float gi = (float)Gs[gb + u], gf = (float)Gs[gb + 512 + u];
      float gg = (float)Gs[gb + 1024 + u], go = (float)Gs[gb + 1536 + u];
      float cc = sigm(gf) * c1r[q] + sigm(gi) * tanhf(gg);
      c1r[q] = cc;
      h1h[e] = (_Float16)(sigm(go) * tanhf(cc));
    }
    __syncthreads();
    // phase 3: G = h1 @ wih2.T + bih2 + h2 @ whh2.T + bhh2
    for (int nt = wid; nt < 128; nt += 32) {
      v8f acc = {};
      for (int k0 = 0; k0 < HIDD; k0 += 32) {
        v16h a1 = frag_f16_rowK(h1h, HIDD, 0, k0, lane);
        v16h b1 = frag_f16_rowK(wih2h, HIDD, nt * 16, k0, lane);
        acc = __builtin_amdgcn_wmma_f32_16x16x32_f16(false, a1, false, b1,
                                                     (short)0, acc, false, false);
        v16h a2 = frag_f16_rowK(h2h, HIDD, 0, k0, lane);
        v16h b2 = frag_f16_rowK(whh2h, HIDD, nt * 16, k0, lane);
        acc = __builtin_amdgcn_wmma_f32_16x16x32_f16(false, a2, false, b2,
                                                     (short)0, acc, false, false);
      }
      int col = nt * 16 + (lane & 15);
      int hi = (lane >> 4) & 1;
      float bv = bih2[col] + bhh2[col];
#pragma unroll
      for (int j = 0; j < 8; ++j) {
        int bb = j + (hi << 3);
        Gs[bb * 2048 + col] = (_Float16)(acc[j] + bv);
      }
    }
    __syncthreads();
    // phase 4: LSTM cell 2; spill h2 (f16) for hoisted frame GEMM
    _Float16* h2p = H2 + (long long)t * 16 * HIDD;
#pragma unroll
    for (int q = 0; q < 8; ++q) {
      int e = (tid << 3) + q;
      int bb = e >> 9, u = e & 511;
      const int gb = bb * 2048;
      float gi = (float)Gs[gb + u], gf = (float)Gs[gb + 512 + u];
      float gg = (float)Gs[gb + 1024 + u], go = (float)Gs[gb + 1536 + u];
      float cc = sigm(gf) * c2r[q] + sigm(gi) * tanhf(gg);
      c2r[q] = cc;
      _Float16 hh = (_Float16)(sigm(go) * tanhf(cc));
      h2h[e] = hh;
      h2p[e] = hh;
    }
    __syncthreads();
  }
}

__global__ void k_voc(const float* __restrict__ mel, const float* __restrict__ vw,
                      const float* __restrict__ vb, float* __restrict__ wav) {
  int idx = blockIdx.x * blockDim.x + threadIdx.x;
  if (idx >= BSZ * TLEN) return;
  int t = idx % TLEN, b = idx / TLEN;
  float s = vb[0];
  for (int c = 0; c < NMEL; ++c) s += mel[((long long)b * NMEL + c) * TLEN + t] * vw[c];
  wav[idx] = s;
}

static inline int cdiv(long long a, long long b) { return (int)((a + b - 1) / b); }

extern "C" void kernel_launch(void* const* d_in, const int* in_sizes, int n_in,
                              void* d_out, int out_size, void* d_ws, size_t ws_size,
                              hipStream_t stream) {
  (void)in_sizes; (void)n_in; (void)out_size; (void)ws_size;
  auto f = [&](int i) { return (const float*)d_in[i]; };

  const float* ref_mel = f(0);
  const int* tok = (const int*)d_in[1];
  const float *s_c1w = f(2), *s_c1b = f(3), *s_c2w = f(4), *s_c2b = f(5);
  const float *s_c3w = f(6), *s_c3b = f(7), *s_fcw = f(8), *s_fcb = f(9);
  const float *s_clw = f(10), *s_clb = f(11);
  const float *e_c1w = f(12), *e_c1b = f(13), *e_c2w = f(14), *e_c2b = f(15);
  const float *e_c3w = f(16), *e_c3b = f(17), *e_fcw = f(18), *e_fcb = f(19);
  const float *e_clw = f(20), *e_clb = f(21);
  const float* emb = f(22);
  const float *linw = f(71), *linb = f(72);
  const float *wih1 = f(73), *whh1 = f(74), *bih1 = f(75), *bhh1 = f(76);
  const float *wih2 = f(77), *whh2 = f(78), *bih2 = f(79), *bhh2 = f(80);
  const float *dfcw = f(81), *dfcb = f(82);
  const float *vocw = f(83), *vocb = f(84);

  float* out = (float*)d_out;
  float* out_mel = out;
  float* out_wav = out + 1024000;
  float* out_spk = out + 1036800;
  float* out_emo = out + 1038848;

  char* ws = (char*)d_ws;
  size_t off = 0;
  auto alloc = [&](size_t bytes) {
    void* p = (void*)(ws + off);
    off += (bytes + 255) & ~(size_t)255;
    return p;
  };
  // f32 buffers
  float* xbuf = (float*)alloc(4096LL * 256 * 4);
  float* tmp  = (float*)alloc(4096LL * 256 * 4);
  float* enc  = (float*)alloc(4096LL * 512 * 4);
  float* gx1  = (float*)alloc(12800LL * 2048 * 4);
  int*   padm = (int*)alloc(4096LL * 4);
  float* styl = (float*)alloc(16LL * SDM * 4);
  float* x1   = (float*)alloc(16LL * 128 * TLEN * 4);
  float* x2   = (float*)alloc(16LL * 256 * TLEN * 4);
  float* x3   = (float*)alloc(16LL * 256 * TLEN * 4);
  // f16 activation buffers
  _Float16* xh    = (_Float16*)alloc(4096LL * 256 * 2);
  _Float16* qkvh  = (_Float16*)alloc(4096LL * 768 * 2);
  _Float16* sch   = (_Float16*)alloc(64LL * 256 * 256 * 2);
  _Float16* obufh = (_Float16*)alloc(4096LL * 256 * 2);
  _Float16* ffnh  = (_Float16*)alloc(4096LL * 1024 * 2);
  _Float16* vT    = (_Float16*)alloc(64LL * 64 * 256 * 2);
  _Float16* xdech = (_Float16*)alloc(12800LL * 1024 * 2);
  _Float16* H2h   = (_Float16*)alloc(12800LL * 512 * 2);
  // f16 weight buffers
  _Float16* whh1h = (_Float16*)alloc(2048LL * 512 * 2);
  _Float16* wih2h = (_Float16*)alloc(2048LL * 512 * 2);
  _Float16* whh2h = (_Float16*)alloc(2048LL * 512 * 2);
  _Float16* wih1h = (_Float16*)alloc(2048LL * 1024 * 2);  // K padded 976->1024
  _Float16* linwh = (_Float16*)alloc(512LL * 256 * 2);
  _Float16* dfcwh = (_Float16*)alloc(80LL * 512 * 2);
  _Float16 *qkvwh[4], *owh[4], *f1wh[4], *f2wh[4];
  for (int l = 0; l < 4; ++l) {
    qkvwh[l] = (_Float16*)alloc(768LL * 256 * 2);
    owh[l]   = (_Float16*)alloc(256LL * 256 * 2);
    f1wh[l]  = (_Float16*)alloc(1024LL * 256 * 2);
    f2wh[l]  = (_Float16*)alloc(256LL * 1024 * 2);
  }

  // ---- one-time weight conversion to f16 ----
  {
    long long n = 2048LL * 512;
    k_f2h<<<cdiv(n, 256), 256, 0, stream>>>(whh1, whh1h, n);
    k_f2h<<<cdiv(n, 256), 256, 0, stream>>>(wih2, wih2h, n);
    k_f2h<<<cdiv(n, 256), 256, 0, stream>>>(whh2, whh2h, n);
    k_f2h_pad<<<cdiv(2048LL * 1024, 256), 256, 0, stream>>>(wih1, wih1h, 2048, 976, 1024);
    k_f2h<<<cdiv(512LL * 256, 256), 256, 0, stream>>>(linw, linwh, 512LL * 256);
    k_f2h<<<cdiv(80LL * 512, 256), 256, 0, stream>>>(dfcw, dfcwh, 80LL * 512);
    for (int l = 0; l < 4; ++l) {
      k_f2h<<<cdiv(768LL * 256, 256), 256, 0, stream>>>(f(23 + 12 * l + 0), qkvwh[l], 768LL * 256);
      k_f2h<<<cdiv(256LL * 256, 256), 256, 0, stream>>>(f(23 + 12 * l + 2), owh[l], 256LL * 256);
      k_f2h<<<cdiv(1024LL * 256, 256), 256, 0, stream>>>(f(23 + 12 * l + 6), f1wh[l], 1024LL * 256);
      k_f2h<<<cdiv(256LL * 1024, 256), 256, 0, stream>>>(f(23 + 12 * l + 8), f2wh[l], 256LL * 1024);
    }
  }

  // ---- reference encoders ----
  long long n1 = (long long)BSZ * 128 * TLEN;
  long long n2 = (long long)BSZ * 256 * TLEN;
  k_conv1<<<cdiv(n1, 256), 256, 0, stream>>>(ref_mel, s_c1w, s_c1b, x1, 3);
  k_conv<<<cdiv(n2, 256), 256, 0, stream>>>(x1, s_c2w, s_c2b, x2, 128, 256, 3);
  k_conv<<<cdiv(n2, 256), 256, 0, stream>>>(x2, s_c3w, s_c3b, x3, 256, 256, 3);
  k_pool_fc<<<16, 256, 0, stream>>>(x3, s_fcw, s_fcb, s_clw, s_clb, styl, out_spk,
                                    256, 128, 0, 0);
  k_conv1<<<cdiv(n1, 256), 256, 0, stream>>>(ref_mel, e_c1w, e_c1b, x1, 5);
  k_conv<<<cdiv(n2, 256), 256, 0, stream>>>(x1, e_c2w, e_c2b, x2, 128, 256, 5);
  k_conv<<<cdiv(n2, 256), 256, 0, stream>>>(x2, e_c3w, e_c3b, x3, 256, 256, 3);
  k_pool_fc<<<16, 256, 0, stream>>>(x3, e_fcw, e_fcb, e_clw, e_clb, styl, out_emo,
                                    128, 8, 256, 1);

  // ---- text transformer ----
  k_embed_pe<<<4096, 256, 0, stream>>>(tok, emb, xbuf, xh, padm);
  for (int l = 0; l < 4; ++l) {
    const float* qkvb = f(23 + 12 * l + 1);
    const float* ob   = f(23 + 12 * l + 3);
    const float* ln1g = f(23 + 12 * l + 4);
    const float* ln1b = f(23 + 12 * l + 5);
    const float* f1b  = f(23 + 12 * l + 7);
    const float* f2b  = f(23 + 12 * l + 9);
    const float* ln2g = f(23 + 12 * l + 10);
    const float* ln2b = f(23 + 12 * l + 11);

    // qkv (f16 out only): [4096,768] = xh @ qkvw.T
    gemm16<0><<<dim3(384), 256, 0, stream>>>(
        xh, qkvwh[l], qkvb, nullptr, qkvh, 4096, 768, 256, 256, 256, 768, 1,
        0, 0, 0, 0, 0, 0);
    // scores = q @ k.T, batched over (b,h)=64
    gemm16<0><<<dim3(8, 64), 256, 0, stream>>>(
        qkvh, qkvh + 256, nullptr, nullptr, sch, 256, 256, 64, 768, 768, 256, 4,
        196608LL, 64LL, 196608LL, 64LL, 262144LL, 65536LL);
    k_softmax_mask<<<16384, 256, 0, stream>>>(sch, padm);
    k_tr_v<<<cdiv(64LL * 64 * 256, 256), 256, 0, stream>>>(qkvh, vT);
    // o = a @ vT.T, batched
    gemm16<0><<<dim3(2, 64), 256, 0, stream>>>(
        sch, vT, nullptr, nullptr, obufh, 256, 64, 256, 256, 256, 256, 4,
        262144LL, 65536LL, 65536LL, 16384LL, 65536LL, 64LL);
    // proj (f32 out for residual)
    gemm16<0><<<dim3(128), 256, 0, stream>>>(
        obufh, owh[l], ob, tmp, nullptr, 4096, 256, 256, 256, 256, 256, 1,
        0, 0, 0, 0, 0, 0);
    k_add_ln<<<4096, 256, 0, stream>>>(xbuf, tmp, ln1g, ln1b, xh);
    // ffn
    gemm16<1><<<dim3(512), 256, 0, stream>>>(
        xh, f1wh[l], f1b, nullptr, ffnh, 4096, 1024, 256, 256, 256, 1024, 1,
        0, 0, 0, 0, 0, 0);
    gemm16<0><<<dim3(128), 256, 0, stream>>>(
        ffnh, f2wh[l], f2b, tmp, nullptr, 4096, 256, 1024, 1024, 1024, 256, 1,
        0, 0, 0, 0, 0, 0);
    k_add_ln<<<4096, 256, 0, stream>>>(xbuf, tmp, ln2g, ln2b, xh);
  }
  // enc = x @ linw.T + linb (f32)
  gemm16<0><<<dim3(256), 256, 0, stream>>>(
      xh, linwh, linb, enc, nullptr, 4096, 512, 256, 256, 256, 512, 1,
      0, 0, 0, 0, 0, 0);

  // ---- decoder ----
  k_build_xdec<<<cdiv(12800LL * 1024, 256), 256, 0, stream>>>(enc, styl, ref_mel, xdech);
  // gx1 = xdec @ wih1.T + bih1 (K padded to 1024 with zeros on both operands)
  gemm16<0><<<dim3(3200), 256, 0, stream>>>(
      xdech, wih1h, bih1, gx1, nullptr, 12800, 2048, 1024, 1024, 1024, 2048, 1,
      0, 0, 0, 0, 0, 0);
  // persistent recurrent scan (single WGP)
  k_scan<<<1, 1024, 0, stream>>>(gx1, whh1h, wih2h, whh2h, bhh1, bih2, bhh2, H2h);
  // mel = H2 @ fcw.T + fcb, stored as [B,NM,T]
  gemm16<2><<<dim3(200), 256, 0, stream>>>(
      H2h, dfcwh, dfcb, out_mel, nullptr, 12800, 80, 512, 512, 512, 800, 1,
      0, 0, 0, 0, 0, 0);
  k_voc<<<cdiv(16LL * 800, 256), 256, 0, stream>>>(out_mel, vocw, vocb, out_wav);
}
